// MultiHeadAttention_28260884808093
// MI455X (gfx1250) — matrix operand order repaired and checked by reading, not compiled, not run on hardware
//
#include <hip/hip_runtime.h>
#include <hip/hip_bf16.h>

typedef _Float16 f16;
typedef __attribute__((ext_vector_type(16))) _Float16 v16h;
typedef __attribute__((ext_vector_type(8)))  float    v8f;
typedef __attribute__((ext_vector_type(4)))  unsigned int u32x4;
typedef __attribute__((ext_vector_type(8)))  int      i32x8;
typedef __attribute__((ext_vector_type(4)))  int      i32x4;

#define B_   4
#define H_   16
#define S_   2048
#define HD   64
#define DMOD 1024
#define NC   (S_ / 32)      // 64 key chunks of 32

// ---------------------------------------------------------------------------
// DPP16 butterfly reductions over the 16-lane half-groups (VALU only, no LDS).
// ---------------------------------------------------------------------------
#if defined(__AMDGCN__) && __has_builtin(__builtin_amdgcn_update_dpp)
template <int CTRL>
__device__ __forceinline__ float dpp_mov(float x) {
    return __int_as_float(__builtin_amdgcn_update_dpp(0, __float_as_int(x), CTRL, 0xF, 0xF, true));
}
__device__ __forceinline__ float rmax16(float x) {
    x = fmaxf(x, dpp_mov<0xB1>(x));    // quad_perm [1,0,3,2]  : xor 1
    x = fmaxf(x, dpp_mov<0x4E>(x));    // quad_perm [2,3,0,1]  : xor 2
    x = fmaxf(x, dpp_mov<0x141>(x));   // row_half_mirror      : combine quads
    x = fmaxf(x, dpp_mov<0x140>(x));   // row_mirror           : combine 8-lane halves
    return x;
}
__device__ __forceinline__ float rsum16(float x) {
    x += dpp_mov<0xB1>(x);
    x += dpp_mov<0x4E>(x);
    x += dpp_mov<0x141>(x);
    x += dpp_mov<0x140>(x);
    return x;
}
#else
__device__ __forceinline__ float rmax16(float x) {
    for (int m = 1; m < 16; m <<= 1) x = fmaxf(x, __shfl_xor(x, m, 32));
    return x;
}
__device__ __forceinline__ float rsum16(float x) {
    for (int m = 1; m < 16; m <<= 1) x += __shfl_xor(x, m, 32);
    return x;
}
#endif

// ---------------------------------------------------------------------------
// Kernel 0: f32 -> f16 streaming convert (used for fc_w, K, V).
// ---------------------------------------------------------------------------
__global__ void cvt_kernel(const float* __restrict__ src, f16* __restrict__ dst, int n) {
    int i = blockIdx.x * blockDim.x + threadIdx.x;
    int stride = gridDim.x * blockDim.x;
    for (; i < n; i += stride) dst[i] = (f16)src[i];
}

// ---------------------------------------------------------------------------
// Kernel 1: fused flash attention per (b, h, 128-row q block).
// 4 waves / block; each wave owns TWO 16-row Q subtiles (32 rows).
// K tiles (pre-converted f16) moved by the Tensor Data Mover, double-buffered
// and overlapped with compute; V tiles (f16) manually transposed. All WMMA
// B-operand fragments are loaded from LDS once and fed to both Q subtiles.
// ---------------------------------------------------------------------------
__launch_bounds__(128)
__global__ void attn_kernel(const float* __restrict__ q, const f16* __restrict__ k16,
                            const f16* __restrict__ v16, f16* __restrict__ att16) {
    __shared__ f16 Klds[2][32 * HD];     // ping-pong [key][dim] f16 (8 KB), TDM dest
    __shared__ f16 Vt[2][HD * 32];       // ping-pong [dim][key] f16 (8 KB)
    __shared__ f16 Pst[4][2][16 * 32];   // per-wave, per-subtile P staging (8 KB)

    const int qblk = blockIdx.x & 15;          // 16 q-blocks of 128 rows
    const int bh   = blockIdx.x >> 4;
    const int b    = bh >> 4;
    const int h    = bh & 15;
    const int wv   = threadIdx.x >> 5;
    const int lane = threadIdx.x & 31;
    const int lm   = lane & 15;
    const int hi   = lane >> 4;
    const int q0   = qblk * 128 + wv * 32;

    // Two Q subtiles in WMMA A layout: aQ[qt][c] covers rows q0+qt*16.., dims c*32..
    v16h aQ[2][2];
    #pragma unroll
    for (int qt = 0; qt < 2; ++qt) {
        const float* qrow = q + (size_t)(b * S_ + q0 + qt * 16 + lm) * DMOD + h * HD;
        #pragma unroll
        for (int j = 0; j < 8; ++j) {
            aQ[qt][0][j]     = (f16)qrow[0  + hi * 8 + j];
            aQ[qt][0][8 + j] = (f16)qrow[16 + hi * 8 + j];
            aQ[qt][1][j]     = (f16)qrow[32 + hi * 8 + j];
            aQ[qt][1][8 + j] = (f16)qrow[48 + hi * 8 + j];
        }
    }

    v8f oacc[2][4];
    #pragma unroll
    for (int qt = 0; qt < 2; ++qt)
        #pragma unroll
        for (int t = 0; t < 4; ++t) oacc[qt][t] = {};
    float mrow[2][8], lrow[2][8];
    #pragma unroll
    for (int qt = 0; qt < 2; ++qt)
        #pragma unroll
        for (int r = 0; r < 8; ++r) { mrow[qt][r] = -3.0e38f; lrow[qt][r] = 0.0f; }

    const float scale = 1.0f / 64.0f;   // reference divides by HEAD_DIM (not sqrt)
    const int tid  = threadIdx.x;
    const int krow = tid >> 2;          // 0..31  (key row within chunk)
    const int dst0 = (tid & 3) * 16;    // 16-dim slice per thread

    // issue K (TDM) + V (manual transpose) loads for chunk kcc into buffer buf
    auto load_tile = [&](int kcc, int buf) {
        const int key0 = kcc * 32;
        {
            const f16* vp = v16 + (size_t)(b * S_ + key0 + krow) * DMOD + h * HD + dst0;
            #pragma unroll
            for (int i = 0; i < 16; ++i)
                Vt[buf][(dst0 + i) * 32 + krow] = vp[i];
            if (kcc + 1 < NC)
                __builtin_prefetch(vp + 32 * DMOD, 0, 0);
        }
#if defined(__AMDGCN__) && __has_builtin(__builtin_amdgcn_tensor_load_to_lds)
        if (threadIdx.x < 32) {
            // ---- Tensor Data Mover: 2D tile 64(x) x 32(y) f16, stride 1024 elems ----
            unsigned long long ga =
                (unsigned long long)(const void*)(k16 + (size_t)(b * S_ + key0) * DMOD + h * HD);
            unsigned ldsaddr =
                (unsigned)(unsigned long long)(__attribute__((address_space(3))) void*)&Klds[buf][0];
            u32x4 g0;
            g0[0] = 1u;                                                  // count=1, user mode
            g0[1] = ldsaddr;                                             // lds_addr
            g0[2] = (unsigned)(ga & 0xFFFFFFFFull);                      // global_addr[31:0]
            g0[3] = (unsigned)((ga >> 32) & 0x01FFFFFFull) | (2u << 30); // ga[56:32] | type=2
            i32x8 g1;
            g1[0] = (int)(1u << 16);        // workgroup_mask=0, data_size=1 (2B)
            g1[1] = (int)(64u << 16);       // tensor_dim0 = 64 elems  (bits 79:48)
            g1[2] = (int)(32u << 16);       // tensor_dim1 = 32        (bits 111:80)
            g1[3] = (int)(64u << 16);       // tile_dim0   = 64        (bits 127:112)
            g1[4] = (int)32u;               // tile_dim1=32, tile_dim2=0
            g1[5] = (int)1024u;             // tensor_dim0_stride = 1024 elems
            g1[6] = 0;
            g1[7] = 0;
            i32x4 gz4 = {};
#if __clang_major__ >= 23
            i32x8 gz8 = {};
            __builtin_amdgcn_tensor_load_to_lds(g0, g1, gz4, gz4, gz8, 0);
#else
            __builtin_amdgcn_tensor_load_to_lds(g0, g1, gz4, gz4, 0);
#endif
        }
#else
        {
            const f16* kp = k16 + (size_t)(b * S_ + key0 + krow) * DMOD + h * HD + dst0;
            #pragma unroll
            for (int i = 0; i < 16; ++i)
                Klds[buf][krow * HD + dst0 + i] = kp[i];
        }
#endif
    };

    // ---- prologue: fill buffer 0 ----
    load_tile(0, 0);
#if defined(__AMDGCN__) && __has_builtin(__builtin_amdgcn_tensor_load_to_lds)
    if (threadIdx.x < 32) __builtin_amdgcn_s_wait_tensorcnt(0);
#endif
    __syncthreads();

    for (int kc = 0; kc < NC; ++kc) {
        const int cur = kc & 1;
        if (kc + 1 < NC) load_tile(kc + 1, cur ^ 1);   // overlap next tile with compute

        // scores: sc[qt][kt] = Q(qt) . K^T(kt), K-dim 64 via two 32-chunks.
        // each bk fragment is loaded once and used by both Q subtiles.
        v8f sc[2][2] = {{{}, {}}, {{}, {}}};
        #pragma unroll
        for (int kt = 0; kt < 2; ++kt) {
            #pragma unroll
            for (int c = 0; c < 2; ++c) {
                v16h bk;
                #pragma unroll
                for (int j = 0; j < 16; ++j)
                    bk[j] = Klds[cur][(kt * 16 + lm) * HD + c * 32 + hi * 8 + (j & 7) + (j & 8) * 2];
                #pragma unroll
                for (int qt = 0; qt < 2; ++qt)
                    sc[qt][kt] = __builtin_amdgcn_wmma_f32_16x16x32_f16(
                        false, aQ[qt][c], false, bk, (short)0, sc[qt][kt], false, false);
            }
        }

        // online softmax per subtile. C-layout: VGPR r <-> row (r+8*hi), lane%16 <-> col.
        #pragma unroll
        for (int qt = 0; qt < 2; ++qt) {
            float corr[8], p0[8], p1[8];
            #pragma unroll
            for (int r = 0; r < 8; ++r) {
                float x0 = sc[qt][0][r] * scale, x1 = sc[qt][1][r] * scale;
                float mx   = rmax16(fmaxf(x0, x1));
                float mnew = fmaxf(mrow[qt][r], mx);
                float e0 = __expf(x0 - mnew), e1 = __expf(x1 - mnew);
                float rs = rsum16(e0 + e1);
                corr[r] = __expf(mrow[qt][r] - mnew);
                lrow[qt][r] = lrow[qt][r] * corr[r] + rs;
                mrow[qt][r] = mnew;
                p0[r] = e0; p1[r] = e1;
            }
            #pragma unroll
            for (int t = 0; t < 4; ++t)
                #pragma unroll
                for (int r = 0; r < 8; ++r) oacc[qt][t][r] *= corr[r];
            // stage P (f16) per-wave/per-subtile in LDS
            #pragma unroll
            for (int r = 0; r < 8; ++r) {
                Pst[wv][qt][(r + 8 * hi) * 32 + lm]      = (f16)p0[r];
                Pst[wv][qt][(r + 8 * hi) * 32 + 16 + lm] = (f16)p1[r];
            }
        }
        asm volatile("s_wait_dscnt 0" ::: "memory");

        // re-read P in WMMA A layout
        v16h ap[2];
        #pragma unroll
        for (int qt = 0; qt < 2; ++qt)
            #pragma unroll
            for (int j = 0; j < 8; ++j) {
                ap[qt][j]     = Pst[wv][qt][lm * 32 + hi * 8 + j];
                ap[qt][8 + j] = Pst[wv][qt][lm * 32 + 16 + hi * 8 + j];
            }

        // O += P(16x32) * V(32x64): 4 dim-tiles; bv loaded once per tile
        #pragma unroll
        for (int t = 0; t < 4; ++t) {
            v16h bv;
            #pragma unroll
            for (int j = 0; j < 16; ++j) bv[j] = Vt[cur][(t * 16 + lm) * 32 + hi * 8 + (j & 7) + (j & 8) * 2];
            #pragma unroll
            for (int qt = 0; qt < 2; ++qt)
                oacc[qt][t] = __builtin_amdgcn_wmma_f32_16x16x32_f16(
                    false, ap[qt], false, bv, (short)0, oacc[qt][t], false, false);
        }

#if defined(__AMDGCN__) && __has_builtin(__builtin_amdgcn_tensor_load_to_lds)
        if (kc + 1 < NC && threadIdx.x < 32) __builtin_amdgcn_s_wait_tensorcnt(0);
#endif
        __syncthreads();
    }

    // normalize rows by 1/l and store f16 attention output (natural [B*S][1024] layout)
    #pragma unroll
    for (int qt = 0; qt < 2; ++qt) {
        float inv[8];
        #pragma unroll
        for (int r = 0; r < 8; ++r) inv[r] = 1.0f / lrow[qt][r];
        #pragma unroll
        for (int t = 0; t < 4; ++t) {
            #pragma unroll
            for (int r = 0; r < 8; ++r) {
                int srow = q0 + qt * 16 + r + 8 * hi;
                att16[(size_t)(b * S_ + srow) * DMOD + h * HD + t * 16 + lm] =
                    (f16)(oacc[qt][t][r] * inv[r]);
            }
        }
    }
}

// ---------------------------------------------------------------------------
// Kernel 2: out[8192,1024] = att16 @ fc_w^T + bias, f16 WMMA, f32 accumulate.
// Each wave: 16x64 output tile, K-loop over 1024 in steps of 32.
// ---------------------------------------------------------------------------
__launch_bounds__(256)
__global__ void proj_kernel(const f16* __restrict__ a16, const f16* __restrict__ w16,
                            const float* __restrict__ bias, float* __restrict__ out) {
    const int wv   = threadIdx.x >> 5;
    const int lane = threadIdx.x & 31;
    const int lm   = lane & 15;
    const int hi   = lane >> 4;
    const int wid  = blockIdx.x * 8 + wv;
    const int mt   = wid >> 4;              // 512 row tiles of 16
    const int nc   = wid & 15;              // 16 col blocks of 64
    const int m0   = mt * 16, n0 = nc * 64;

    v8f acc[4];
    #pragma unroll
    for (int t = 0; t < 4; ++t) {
        float bval = bias[n0 + t * 16 + lm];       // bias depends only on column
        #pragma unroll
        for (int r = 0; r < 8; ++r) acc[t][r] = bval;
    }

    const f16* arow = a16 + (size_t)(m0 + lm) * DMOD;
    for (int kk = 0; kk < DMOD; kk += 32) {
        v16h a;
        #pragma unroll
        for (int j = 0; j < 8; ++j) {
            a[j]     = arow[kk + hi * 8 + j];
            a[8 + j] = arow[kk + 16 + hi * 8 + j];
        }
        #pragma unroll
        for (int t = 0; t < 4; ++t) {
            const f16* wrow = w16 + (size_t)(n0 + t * 16 + lm) * DMOD + kk + hi * 8;
            v16h bw;
            #pragma unroll
            for (int j = 0; j < 16; ++j) bw[j] = wrow[(j & 7) + (j & 8) * 2];
            acc[t] = __builtin_amdgcn_wmma_f32_16x16x32_f16(false, a, false, bw, (short)0, acc[t], false, false);
        }
    }

    #pragma unroll
    for (int t = 0; t < 4; ++t)
        #pragma unroll
        for (int r = 0; r < 8; ++r)
            out[(size_t)(m0 + r + 8 * hi) * DMOD + n0 + t * 16 + lm] = acc[t][r];
}

// ---------------------------------------------------------------------------
extern "C" void kernel_launch(void* const* d_in, const int* in_sizes, int n_in,
                              void* d_out, int out_size, void* d_ws, size_t ws_size,
                              hipStream_t stream) {
    (void)in_sizes; (void)n_in; (void)out_size; (void)ws_size;
    const float* q   = (const float*)d_in[0];
    const float* k   = (const float*)d_in[1];
    const float* v   = (const float*)d_in[2];
    const float* fcw = (const float*)d_in[3];
    const float* fcb = (const float*)d_in[4];
    float* out = (float*)d_out;

    // workspace layout (f16):
    //   [att16: 8M elems = 16 MB][w16: 1M = 2 MB][k16: 8M = 16 MB][v16: 8M = 16 MB]
    const size_t NQKV = (size_t)B_ * S_ * DMOD;     // 8388608
    f16* att16 = (f16*)d_ws;
    f16* w16   = att16 + NQKV;
    f16* k16   = w16 + (size_t)DMOD * DMOD;
    f16* v16   = k16 + NQKV;

    cvt_kernel<<<512, 256, 0, stream>>>(fcw, w16, DMOD * DMOD);
    cvt_kernel<<<2048, 256, 0, stream>>>(k, k16, (int)NQKV);
    cvt_kernel<<<2048, 256, 0, stream>>>(v, v16, (int)NQKV);
    attn_kernel<<<B_ * H_ * (S_ / 128), 128, 0, stream>>>(q, k16, v16, att16);
    proj_kernel<<<(512 * 16) / 8, 256, 0, stream>>>(att16, w16, fcb, out);
}
